// AsyncConv_50019189129827
// MI455X (gfx1250) — matrix operand order repaired and checked by reading, not compiled, not run on hardware
//
#include <hip/hip_runtime.h>
#include <hip/hip_bf16.h>

typedef __attribute__((ext_vector_type(16))) _Float16 v16h;
typedef __attribute__((ext_vector_type(8)))  _Float16 v8h;
typedef __attribute__((ext_vector_type(4)))  _Float16 v4h;
typedef __attribute__((ext_vector_type(8)))  float    v8f;

#define GB     2
#define GNV    20000
#define GC     64
#define GNR    3
#define GND    16
#define GNF    128
#define GK     3072         // 3*16*64
#define KCH    96           // GK / 32
#define VPB    8            // vertices per block
#define RSTR   72           // 64 data + 8 pad (f16) per (ring,dir) row
#define ROWS   48           // GNR*GND rows per vertex

#if __has_builtin(__builtin_amdgcn_sched_group_barrier)
#define SCHED_GROUP(mask, size, id) __builtin_amdgcn_sched_group_barrier(mask, size, id)
#else
#define SCHED_GROUP(mask, size, id)
#endif

// ---------------------------------------------------------------------------
// Pack weights f32 (3072 x 128 row-major, K-major == HWIO flat) into f16
// WMMA-B fragments: ws[((kc*8 + nt)*32 + lane)*16 + i]
//   k = kc*32 + (lane>=16 ? 8 : 0) + (i&7) + (i>=8 ? 16 : 0) ;  n = nt*16 + (lane&15)
// ---------------------------------------------------------------------------
__global__ __launch_bounds__(256) void pack_w_kernel(
    const float* __restrict__ kern, _Float16* __restrict__ wsB) {
  int gid  = blockIdx.x * 256 + threadIdx.x;      // 0 .. 393215
  int i    = gid & 15;
  int lane = (gid >> 4) & 31;
  int nt   = (gid >> 9) & 7;
  int kc   = gid >> 12;                           // 0 .. 95
  int k    = kc * 32 + ((lane >> 4) << 3) + (i & 7) + ((i >> 3) << 4);
  int n    = nt * 16 + (lane & 15);
  wsB[gid] = (_Float16)kern[k * GNF + n];
}

// ---------------------------------------------------------------------------
// Main fused gather + rotated-conv GEMM + bias/relu/max kernel.
// Block: 256 threads (8 waves). Wave w owns N-tile w, loops 8 vertices.
// ---------------------------------------------------------------------------
__global__ __launch_bounds__(256) void geo_conv_kernel(
    const float* __restrict__ y,       // (B, NV, C) f32
    const int*   __restrict__ emap,    // (NV, 3, 16) i32
    const v16h*  __restrict__ wsB,     // packed B fragments (f16)
    const float* __restrict__ bias,    // (NF,) f32
    float*       __restrict__ out) {   // (B, NV, NF) f32

  __shared__ _Float16 gb[VPB * ROWS * RSTR];      // 55,296 bytes

  const int t   = threadIdx.x;
  const int bv0 = blockIdx.x * VPB;

  // ---- Gather stage: 8 vertices * 48 rows * 16 float4 = 6144 float4 ----
  #pragma unroll 4
  for (int it = 0; it < 24; ++it) {
    int lin  = t + it * 256;           // 0 .. 6143
    int row  = lin >> 4;               // 0 .. 383
    int c4   = lin & 15;
    int vloc = row / ROWS;
    int rd   = row - vloc * ROWS;
    int bv   = bv0 + vloc;
    int b    = bv / GNV;
    int v    = bv - b * GNV;
    int u    = emap[v * ROWS + rd];
    const float4 f = *(const float4*)(y + ((size_t)(b * GNV + u)) * GC + c4 * 4);
    v4h h = { (_Float16)f.x, (_Float16)f.y, (_Float16)f.z, (_Float16)f.w };
    *(v4h*)(gb + (vloc * ROWS + rd) * RSTR + c4 * 4) = h;
  }
  __syncthreads();

  const int w    = t >> 5;             // wave id == N-tile
  const int lane = t & 31;
  const int Mrow = lane & 15;          // dout row this lane owns in A
  const int half = lane >> 4;

  v8f acc[VPB] = {};

  // Software-pipelined B fragment: bnext is fetched for chunk kc+1 while
  // chunk kc's WMMAs consume bfrag.
  v16h bfrag = wsB[(size_t)w * 32 + lane];

  for (int kc = 0; kc < KCH; ++kc) {
    int kn = (kc + 1 < KCH) ? kc + 1 : kc;
    v16h bnext = wsB[(kn * 8 + w) * 32 + lane];

    int rd2 = kc >> 1;                 // (ring*16 + dk)
    int c0  = (kc & 1) << 5;           // 0 or 32
    int r   = rd2 >> 4;
    int dk  = rd2 & 15;
    int srow = r * GND + ((Mrow + dk) & 15);     // rotated direction row
    int coff = c0 + half * 8;

    // Phase 1: all 16 LDS b128 loads for this chunk
    v8h lo[VPB], hi[VPB];
    #pragma unroll
    for (int m = 0; m < VPB; ++m) {
      const _Float16* p = gb + (m * ROWS + srow) * RSTR + coff;
      lo[m] = *(const v8h*)(p);         // K local {0..7} / {8..15}
      hi[m] = *(const v8h*)(p + 16);    // K local {16..23} / {24..31}
    }
    // Phase 2: 8 back-to-back WMMAs on independent accumulators
    #pragma unroll
    for (int m = 0; m < VPB; ++m) {
      v16h a = __builtin_shufflevector(lo[m], hi[m],
               0, 1, 2, 3, 4, 5, 6, 7, 8, 9, 10, 11, 12, 13, 14, 15);
      acc[m] = __builtin_amdgcn_wmma_f32_16x16x32_f16(
                   false, a, false, bfrag, (short)0, acc[m], false, false);
    }
    bfrag = bnext;

    // Enforce pipeline shape: prefetch B (2 vmem reads), then all 16 DS
    // reads (forces distinct A regs -> partial dscnt waits), then 8 WMMAs.
    SCHED_GROUP(0x020, 2, 0);   // VMEM read: bnext
    SCHED_GROUP(0x100, 16, 0);  // DS read:   A fragments
    SCHED_GROUP(0x008, 8, 0);   // MFMA/WMMA
  }

  // ---- Epilogue: max over dout (M), + bias, relu, store ----
  const float bb = bias[w * 16 + (lane & 15)];

  #pragma unroll
  for (int m = 0; m < VPB; ++m) {
    float mx = acc[m][0];
    #pragma unroll
    for (int j = 1; j < 8; ++j) mx = fmaxf(mx, acc[m][j]);
    float other = __shfl_xor(mx, 16, 32);        // merge M 0..7 with 8..15
    float res = fmaxf(fmaxf(mx, other) + bb, 0.0f);
    if (lane < 16) {
      int bv = bv0 + m;
      out[(size_t)bv * GNF + w * 16 + lane] = res;
    }
  }
}

extern "C" void kernel_launch(void* const* d_in, const int* in_sizes, int n_in,
                              void* d_out, int out_size, void* d_ws, size_t ws_size,
                              hipStream_t stream) {
  const float* y    = (const float*)d_in[0];   // (2, 20000, 64)
  const int*   emap = (const int*)  d_in[1];   // (20000, 3, 16)
  const float* kern = (const float*)d_in[2];   // (3, 16, 64, 128)
  const float* bias = (const float*)d_in[3];   // (128,)
  float* out = (float*)d_out;

  // Pack weights: 96*8*32*16 = 393,216 f16 elements (786,432 bytes in d_ws)
  pack_w_kernel<<<1536, 256, 0, stream>>>(kern, (_Float16*)d_ws);

  // 40000 (b,v) pairs / 8 vertices per block = 5000 blocks
  geo_conv_kernel<<<(GB * GNV) / VPB, 256, 0, stream>>>(
      y, emap, (const v16h*)d_ws, bias, out);
}
